// SelfAttention3D_75144747811056
// MI455X (gfx1250) — matrix operand order repaired and checked by reading, not compiled, not run on hardware
//
#include <hip/hip_runtime.h>
#include <hip/hip_bf16.h>

#define BATCH 4
#define SEQ   4096
#define FEAT  256

typedef __attribute__((ext_vector_type(16))) __bf16 v16bf;
typedef __attribute__((ext_vector_type(8)))  __bf16 v8bf;
typedef __attribute__((ext_vector_type(8)))  float  v8f;
typedef __attribute__((ext_vector_type(4)))  unsigned int v4u;
typedef __attribute__((ext_vector_type(8)))  int    v8i;
typedef __attribute__((ext_vector_type(4)))  int    v4i;

static __device__ __forceinline__ v16bf join8(v8bf lo, v8bf hi) {
  v16bf r;
#pragma unroll
  for (int i = 0; i < 8; ++i) { r[i] = lo[i]; r[8 + i] = hi[i]; }
  return r;
}

// ---------------------------------------------------------------------------
// TDM: async 2-D tile load (global -> LDS), bf16 elements.
// D# packed per CDNA5 ISA 8.3/8.4:
//   group0: count=1 | lds_addr[63:32] | global_addr[120:64] | type=2 [127:126]
//   group1: data_size=1(2B)[17:16] | tensor_dim0[79:48] | tensor_dim1[111:80]
//           tile_dim0[127:112] | tile_dim1[143:128] | dim0_stride[207:160]
// Groups 2/3 zero (2-D tensor). Tracked by TENSORcnt, in-order per wave.
// ---------------------------------------------------------------------------
static __device__ __forceinline__ void tdm_load_2d_bf16(
    unsigned lds_addr, unsigned long long gaddr,
    unsigned tensor_d0, unsigned tensor_d1,
    unsigned tile_d0, unsigned tile_d1,
    unsigned long long stride0_elems)
{
  v4u g0;
  g0[0] = 1u;                                               // count=1, user D#
  g0[1] = lds_addr;                                         // LDS byte address
  g0[2] = (unsigned)(gaddr & 0xffffffffu);                  // global addr lo
  g0[3] = (unsigned)((gaddr >> 32) & 0x01ffffffu) | (2u << 30);  // hi | type=2
  v8i g1;
  g1[0] = (int)(1u << 16);                                  // data_size = 2B
  g1[1] = (int)((tensor_d0 & 0xffffu) << 16);
  g1[2] = (int)((tensor_d0 >> 16) | ((tensor_d1 & 0xffffu) << 16));
  g1[3] = (int)((tensor_d1 >> 16) | (tile_d0 << 16));
  g1[4] = (int)tile_d1;                                     // tile_dim2 = 0
  g1[5] = (int)(unsigned)(stride0_elems & 0xffffffffu);
  g1[6] = (int)(unsigned)((stride0_elems >> 32) & 0xffffu); // dim1_stride = 0
  g1[7] = 0;
  v4i z4 = {0, 0, 0, 0};
#if __clang_major__ >= 23
  v8i z8 = {0, 0, 0, 0, 0, 0, 0, 0};
  __builtin_amdgcn_tensor_load_to_lds(g0, g1, z4, z4, z8, 0);
#else
  __builtin_amdgcn_tensor_load_to_lds(g0, g1, z4, z4, 0);
#endif
}

static __device__ __forceinline__ unsigned lds_off(const void* p) {
  // generic -> LDS byte offset: LDS aperture maps addr[31:0] to LDS space
  return (unsigned)(unsigned long long)(uintptr_t)p;
}

// ---------------------------------------------------------------------------
// Kernel 1: QKV projection.  out = x @ W^T + b as 16x16 f32 tiles via
// V_WMMA_F32_16X16X32_BF16 over K=256 (8 WMMAs). Q,K row-major bf16; V stored
// transposed [B][F][N] so the PV GEMM B-operand is contiguous per lane.
// ---------------------------------------------------------------------------
__global__ __launch_bounds__(32)
void qkv_proj_kernel(const float* __restrict__ x,
                     const float* __restrict__ Wq, const float* __restrict__ bq,
                     const float* __restrict__ Wk, const float* __restrict__ bk,
                     const float* __restrict__ Wv, const float* __restrict__ bv,
                     unsigned short* __restrict__ Qo,
                     unsigned short* __restrict__ Ko,
                     unsigned short* __restrict__ Vto)
{
  const int mt    = blockIdx.x;
  const int nt    = blockIdx.y;
  const int which = blockIdx.z;
  const int lane  = threadIdx.x;
  const int hf    = lane >> 4;
  const int lrow  = lane & 15;

  const float* W    = (which == 0) ? Wq : (which == 1) ? Wk : Wv;
  const float* bias = (which == 0) ? bq : (which == 1) ? bk : bv;

  __bf16* Q  = (__bf16*)Qo;
  __bf16* K  = (__bf16*)Ko;
  __bf16* Vt = (__bf16*)Vto;

  v8f acc = {};
#pragma unroll
  for (int k0 = 0; k0 < FEAT; k0 += 32) {
    const float* xr = x + (size_t)(mt * 16 + lrow) * FEAT + k0 + hf * 16;
    v16bf a, b;
#pragma unroll
    for (int i = 0; i < 8; ++i) { a[i] = (__bf16)xr[i]; a[8 + i] = (__bf16)xr[8 + i]; }
    const float* wr = W + (size_t)(nt * 16 + lrow) * FEAT + k0 + hf * 16;
#pragma unroll
    for (int i = 0; i < 16; ++i) b[i] = (__bf16)wr[i];
    acc = __builtin_amdgcn_wmma_f32_16x16x32_bf16(false, a, false, b, (short)0, acc,
                                                  false, false);
  }

  const int   col = nt * 16 + lrow;
  const float bc  = bias[col];
#pragma unroll
  for (int r = 0; r < 8; ++r) {
    const int   rowg = mt * 16 + r + 8 * hf;
    const float v    = acc[r] + bc;
    if (which == 0) {
      Q[(size_t)rowg * FEAT + col] = (__bf16)v;
    } else if (which == 1) {
      K[(size_t)rowg * FEAT + col] = (__bf16)v;
    } else {
      const int bidx = rowg >> 12;
      const int nidx = rowg & (SEQ - 1);
      Vt[((size_t)bidx * FEAT + col) * SEQ + nidx] = (__bf16)v;
    }
  }
}

// ---------------------------------------------------------------------------
// Kernel 2: flash attention, TDM double-buffered. One wave per 16-query tile.
// Per 32-key block: two tensor_load_to_lds (K 32x256, V^T 256x32, 16 KB each)
// for block i+1 issued before computing block i from LDS; s_wait_tensorcnt 2
// guarantees block i's pair is complete (tensor ops are in-order per wave).
// ---------------------------------------------------------------------------
__global__ __launch_bounds__(32)
void attn_kernel(const unsigned short* __restrict__ Qi,
                 const unsigned short* __restrict__ Ki,
                 const unsigned short* __restrict__ Vti,
                 float* __restrict__ out)
{
  __shared__ __align__(16) __bf16 qs[16 * FEAT];        // 8 KB query tile
  __shared__ __align__(16) float  ps[16 * 32];          // 2 KB P staging
  __shared__ __align__(16) __bf16 ks[2][32 * FEAT];     // 2 x 16 KB K tiles
  __shared__ __align__(16) __bf16 vs[2][FEAT * 32];     // 2 x 16 KB V^T tiles

  const int lane = threadIdx.x;
  const int hf   = lane >> 4;
  const int lrow = lane & 15;
  const int qt   = blockIdx.x;
  const int b    = blockIdx.y;

  const __bf16* Qb = (const __bf16*)Qi + ((size_t)b * SEQ + qt * 16) * FEAT;
  const __bf16* Kb = (const __bf16*)Ki + (size_t)b * SEQ * FEAT;
  const __bf16* Vb = (const __bf16*)Vti + (size_t)b * FEAT * SEQ;

  // kick off TDM for key block 0 into buffer 0
  tdm_load_2d_bf16(lds_off(&ks[0][0]),
                   (unsigned long long)(uintptr_t)(Kb),
                   FEAT, SEQ, FEAT, 32, FEAT);
  tdm_load_2d_bf16(lds_off(&vs[0][0]),
                   (unsigned long long)(uintptr_t)(Vb),
                   SEQ, FEAT, 32, FEAT, SEQ);

  // stage Q tile
#pragma unroll
  for (int c = 0; c < 128; c += 8)
    *(v8bf*)(qs + lrow * FEAT + hf * 128 + c) =
        *(const v8bf*)(Qb + lrow * FEAT + hf * 128 + c);
  __syncthreads();

  v8f accum[16] = {};
  float mrow[8], lsum[8];
#pragma unroll
  for (int r = 0; r < 8; ++r) { mrow[r] = -3.0e38f; lsum[r] = 0.0f; }

  const float sc = 0.0625f;  // 1/sqrt(256)

  for (int kb = 0; kb < SEQ; kb += 32) {
    const int cur = (kb >> 5) & 1;
    const bool more = (kb + 32) < SEQ;
    if (more) {
      // prefetch next key block into the other buffer
      tdm_load_2d_bf16(lds_off(&ks[cur ^ 1][0]),
                       (unsigned long long)(uintptr_t)(Kb + (size_t)(kb + 32) * FEAT),
                       FEAT, SEQ, FEAT, 32, FEAT);
      tdm_load_2d_bf16(lds_off(&vs[cur ^ 1][0]),
                       (unsigned long long)(uintptr_t)(Vb + (kb + 32)),
                       SEQ, FEAT, 32, FEAT, SEQ);
      __builtin_amdgcn_s_wait_tensorcnt(2);   // current pair complete
    } else {
      __builtin_amdgcn_s_wait_tensorcnt(0);
    }
    const __bf16* kt = &ks[cur][0];
    const __bf16* vt = &vs[cur][0];

    // ---- scores: S[16x32] as two 16x16 tiles over K=256 reduction ----
    v8f s0 = {}, s1 = {};
#pragma unroll
    for (int kc = 0; kc < FEAT; kc += 32) {
      const __bf16* qr = qs + lrow * FEAT + kc + hf * 16;
      v16bf a = join8(*(const v8bf*)qr, *(const v8bf*)(qr + 8));
      const __bf16* k0p = kt + (size_t)lrow * FEAT + kc + hf * 16;
      v16bf b0 = join8(*(const v8bf*)k0p, *(const v8bf*)(k0p + 8));
      const __bf16* k1p = k0p + 16 * FEAT;
      v16bf b1 = join8(*(const v8bf*)k1p, *(const v8bf*)(k1p + 8));
      s0 = __builtin_amdgcn_wmma_f32_16x16x32_bf16(false, a, false, b0, (short)0, s0,
                                                   false, false);
      s1 = __builtin_amdgcn_wmma_f32_16x16x32_bf16(false, a, false, b1, (short)0, s1,
                                                   false, false);
    }

    // ---- online softmax; C-layout row r+8*hf lives in VGPR r, col = lane%16.
    // xor masks 1..8 keep the reduction inside each 16-lane half, so per-row
    // state aligns exactly with (register, half) ownership.
    float p0[8], p1[8], corr[8];
#pragma unroll
    for (int r = 0; r < 8; ++r) {
      float mx = fmaxf(s0[r] * sc, s1[r] * sc);
#pragma unroll
      for (int m = 1; m <= 8; m <<= 1) mx = fmaxf(mx, __shfl_xor(mx, m, 32));
      const float mnew = fmaxf(mrow[r], mx);
      corr[r] = __expf(mrow[r] - mnew);
      p0[r]   = __expf(s0[r] * sc - mnew);
      p1[r]   = __expf(s1[r] * sc - mnew);
      float rs = p0[r] + p1[r];
#pragma unroll
      for (int m = 1; m <= 8; m <<= 1) rs += __shfl_xor(rs, m, 32);
      lsum[r] = lsum[r] * corr[r] + rs;
      mrow[r] = mnew;
    }
#pragma unroll
    for (int t = 0; t < 16; ++t)
#pragma unroll
      for (int r = 0; r < 8; ++r) accum[t][r] *= corr[r];

    // ---- C-layout P -> A-layout bf16 via LDS ----
#pragma unroll
    for (int r = 0; r < 8; ++r) {
      ps[(r + 8 * hf) * 32 + lrow]      = p0[r];
      ps[(r + 8 * hf) * 32 + 16 + lrow] = p1[r];
    }
    __syncthreads();
    const float* pr = ps + lrow * 32 + hf * 16;
    v16bf ap;
#pragma unroll
    for (int i = 0; i < 8; ++i) { ap[i] = (__bf16)pr[i]; ap[8 + i] = (__bf16)pr[8 + i]; }
    __syncthreads();

    // ---- accum += P(16x32) @ V(32x16) for 16 d-tiles, V from LDS ----
#pragma unroll
    for (int t = 0; t < 16; ++t) {
      const __bf16* vp = vt + (size_t)(t * 16 + lrow) * 32 + hf * 16;
      v16bf bv = join8(*(const v8bf*)vp, *(const v8bf*)(vp + 8));
      accum[t] = __builtin_amdgcn_wmma_f32_16x16x32_bf16(false, ap, false, bv, (short)0,
                                                         accum[t], false, false);
    }
  }

  // ---- epilogue: normalize by row sums, store f32 ----
  float inv[8];
#pragma unroll
  for (int r = 0; r < 8; ++r) inv[r] = 1.0f / lsum[r];
  float* ob = out + ((size_t)b * SEQ + qt * 16) * FEAT;
#pragma unroll
  for (int t = 0; t < 16; ++t)
#pragma unroll
    for (int r = 0; r < 8; ++r)
      ob[(size_t)(r + 8 * hf) * FEAT + t * 16 + lrow] = accum[t][r] * inv[r];
}

// ---------------------------------------------------------------------------
extern "C" void kernel_launch(void* const* d_in, const int* in_sizes, int n_in,
                              void* d_out, int out_size, void* d_ws, size_t ws_size,
                              hipStream_t stream) {
  const float* x  = (const float*)d_in[0];
  const float* Wq = (const float*)d_in[1];
  const float* bq = (const float*)d_in[2];
  const float* Wk = (const float*)d_in[3];
  const float* bk = (const float*)d_in[4];
  const float* Wv = (const float*)d_in[5];
  const float* bv = (const float*)d_in[6];
  float* out = (float*)d_out;

  const size_t elems = (size_t)BATCH * SEQ * FEAT;
  unsigned short* Qw  = (unsigned short*)d_ws;
  unsigned short* Kw  = Qw + elems;
  unsigned short* Vtw = Kw + elems;

  dim3 pgrid(BATCH * SEQ / 16, FEAT / 16, 3);
  qkv_proj_kernel<<<pgrid, dim3(32), 0, stream>>>(x, Wq, bq, Wk, bk, Wv, bv,
                                                  Qw, Kw, Vtw);

  dim3 agrid(SEQ / 16, BATCH);
  attn_kernel<<<agrid, dim3(32), 0, stream>>>(Qw, Kw, Vtw, out);
}
